// baseModule_14199161880872
// MI455X (gfx1250) — matrix operand order repaired and checked by reading, not compiled, not run on hardware
//
#include <hip/hip_runtime.h>
#include <hip/hip_bf16.h>
#include <math.h>

typedef __attribute__((ext_vector_type(16))) _Float16 v16h;
typedef __attribute__((ext_vector_type(8)))  _Float16 h8;
typedef __attribute__((ext_vector_type(8)))  float    v8f;

#define N_RAYS    4096
#define N_SAMPLES 128

// LDS activation row stride (halfs). 264*2=528 bytes; 528/4=132 = 4 (mod 64)
// -> row m hits bank group 4m: A-fragment ds_load_b128 is bank-conflict-free.
#define ASTRIDE   264
#define BUFHALFS  (32 * ASTRIDE)          // 8448 halfs = 16896 B per buffer

// ---------------- workspace layout -----------------------------------------
// halfs:  W1T[256][64]  @ 0        (16384)
//         W2T[256][256] @ 16384    (65536)
//         W3T[16][256]  @ 81920    (4096)
//         Wc1T[64][64]  @ 86016    (4096)
//         Wc2T[16][64]  @ 90112    (1024)   -> 91136 halfs = 182272 bytes
// bytes:  rgbsig float4[N_RAYS*N_SAMPLES] @ 196608  (8 MB)
#define OFF_W1T  0
#define OFF_W2T  16384
#define OFF_W3T  81920
#define OFF_WC1T 86016
#define OFF_WC2T 90112
#define N_PACK   91136
#define OFF_RGBSIG_BYTES 196608

__device__ inline void lds_fence() {
  asm volatile("s_wait_dscnt 0" ::: "memory");
}

// ---------------- weight prepack: f32 row-major [K][N] -> f16 [N][Kpad] ----
__global__ void prepack_weights(const float* __restrict__ W1, const float* __restrict__ W2,
                                const float* __restrict__ W3, const float* __restrict__ Wc1,
                                const float* __restrict__ Wc2, _Float16* __restrict__ wsh)
{
  int tid = blockIdx.x * blockDim.x + threadIdx.x;
  if (tid >= N_PACK) return;
  float v;
  if (tid < OFF_W2T)       { int n = tid >> 6, k = tid & 63;                 v = (k < 63) ? W1[k * 256 + n] : 0.0f; }
  else if (tid < OFF_W3T)  { int u = tid - OFF_W2T;  int n = u >> 8, k = u & 255; v = W2[k * 256 + n]; }
  else if (tid < OFF_WC1T) { int u = tid - OFF_W3T;  int n = u >> 8, k = u & 255; v = W3[k * 16 + n]; }
  else if (tid < OFF_WC2T) { int u = tid - OFF_WC1T; int n = u >> 6, k = u & 63;  v = (k < 42) ? Wc1[k * 64 + n] : 0.0f; }
  else                     { int u = tid - OFF_WC2T; int n = u >> 6, k = u & 63;  v = (n < 3)  ? Wc2[k * 3 + n]  : 0.0f; }
  wsh[tid] = (_Float16)v;
}

// ---------------- shared ray setup -----------------------------------------
__device__ inline void ray_setup(const float* __restrict__ ro, const float* __restrict__ rd,
                                 int ray, float& dx, float& dy, float& dz,
                                 float& ox, float& oy, float& oz,
                                 float& near, float& dt)
{
  ox = ro[ray * 3 + 0]; oy = ro[ray * 3 + 1]; oz = ro[ray * 3 + 2];
  float vx = rd[ray * 3 + 0], vy = rd[ray * 3 + 1], vz = rd[ray * 3 + 2];
  float inv = rsqrtf(vx * vx + vy * vy + vz * vz);
  dx = vx * inv; dy = vy * inv; dz = vz * inv;
  float nmin = -1e30f, fmax_ = 1e30f;
  float dc[3] = {dx, dy, dz}, oc[3] = {ox, oy, oz};
#pragma unroll
  for (int c = 0; c < 3; c++) {
    float d  = dc[c];
    float dd = (fabsf(d) < 1e-8f) ? 1e-8f : d;
    float id = 1.0f / dd;
    float t0 = (-1.0f - oc[c]) * id;
    float t1 = ( 1.0f - oc[c]) * id;
    nmin  = fmaxf(nmin,  fminf(t0, t1));
    fmax_ = fminf(fmax_, fmaxf(t0, t1));
  }
  near = fmaxf(nmin, 0.02f);
  float far = fminf(fmax_, 1000.0f);
  far = fmaxf(far, near + 0.001f);
  dt = (far - near) * (1.0f / (float)N_SAMPLES);
}

// ---------------- WMMA fragment loaders ------------------------------------
// A: 16x32 f16 tile from LDS act buffer [32][ASTRIDE], row base m0.
// lanes 0-15 : M=lane,  K = {k0+0..7, k0+16..23}
// lanes 16-31: M=lane-16, K = {k0+8..15, k0+24..31}
__device__ inline v16h load_a(const _Float16* __restrict__ base, int m0, int k0, int lane)
{
  int m = (lane & 15) + m0, hf = lane >> 4;
  const _Float16* p = base + m * ASTRIDE + k0 + hf * 8;
  h8 lo = *(const h8*)p;
  h8 hi = *(const h8*)(p + 16);
  return __builtin_shufflevector(lo, hi, 0,1,2,3,4,5,6,7,8,9,10,11,12,13,14,15);
}

// B: 32x16 f16 tile from prepacked [N][Kp] (K contiguous per column).
// lanes 0-15: N=n0+lane, K={k0+0..7, k0+16..23} ; lanes 16-31: N=n0+lane-16, K={k0+8..15, k0+24..31}
__device__ inline v16h load_b(const _Float16* __restrict__ wt, int Kp, int n0, int k0, int lane)
{
  int n  = n0 + (lane & 15);
  int kb = k0 + ((lane >> 4) << 3);
  const _Float16* p = wt + n * Kp + kb;
  h8 lo = *(const h8*)p;
  h8 hi = *(const h8*)(p + 16);
  return __builtin_shufflevector(lo, hi, 0,1,2,3,4,5,6,7,8,9,10,11,12,13,14,15);
}

__device__ inline v8f wmma_f16(v16h a, v16h b, v8f c)
{
  return __builtin_amdgcn_wmma_f32_16x16x32_f16(false, a, false, b, (short)0, c, false, false);
}

// fused layer on a 32-row tile: out = act(in[32xK] @ W[KxN] + b), one wave.
// Each B fragment is loaded once and used by two WMMAs (rows 0-15 and 16-31).
__device__ inline void mlp_layer32(const _Float16* __restrict__ in_lds,
                                   _Float16* __restrict__ out_lds,
                                   const _Float16* __restrict__ wt,
                                   const float* __restrict__ bias,
                                   int K, int N, int lane, bool relu)
{
  int hf = lane >> 4, nn = lane & 15;
  for (int n0 = 0; n0 < N; n0 += 16) {
    v8f acc0 = {}, acc1 = {};
    for (int k0 = 0; k0 < K; k0 += 32) {
      v16h B  = load_b(wt, K, n0, k0, lane);
      v16h A0 = load_a(in_lds, 0,  k0, lane);
      v16h A1 = load_a(in_lds, 16, k0, lane);
      acc0 = wmma_f16(A0, B, acc0);
      acc1 = wmma_f16(A1, B, acc1);
    }
    float bv = bias[n0 + nn];
#pragma unroll
    for (int r = 0; r < 8; r++) {
      float x0 = acc0[r] + bv;                 // C/D layout: lane->N, VGPR r->M (hf*8+r)
      float x1 = acc1[r] + bv;
      if (relu) { x0 = fmaxf(x0, 0.0f); x1 = fmaxf(x1, 0.0f); }
      out_lds[(r + hf * 8) * ASTRIDE + n0 + nn]      = (_Float16)x0;
      out_lds[(16 + r + hf * 8) * ASTRIDE + n0 + nn] = (_Float16)x1;
    }
  }
}

// ---------------- fused NeRF MLP kernel: 1 WG = 1 ray, 1 wave = 32 samples --
__global__ void __launch_bounds__(128) render_kernel(
    const float* __restrict__ rays_o, const float* __restrict__ rays_d,
    const float* __restrict__ b1, const float* __restrict__ b2,
    const float* __restrict__ b3, const float* __restrict__ bc1,
    const float* __restrict__ bc2, const _Float16* __restrict__ wsh,
    float* __restrict__ rgbsig)
{
  extern __shared__ _Float16 smem[];              // 4 waves x 2 x [32][264] = 132 KB
  int lane  = threadIdx.x & 31;
  int wslot = threadIdx.x >> 5;
  int ray   = blockIdx.x;
  int base  = wslot * 32;                         // first sample of this wave
  _Float16* bufA = smem + wslot * (2 * BUFHALFS);
  _Float16* bufB = bufA + BUFHALFS;
  const _Float16* W1T  = wsh + OFF_W1T;
  const _Float16* W2T  = wsh + OFF_W2T;
  const _Float16* W3T  = wsh + OFF_W3T;
  const _Float16* Wc1T = wsh + OFF_WC1T;
  const _Float16* Wc2T = wsh + OFF_WC2T;

  float dx, dy, dz, ox, oy, oz, near, dt;
  ray_setup(rays_o, rays_d, ray, dx, dy, dz, ox, oy, oz, near, dt);

  int hf = lane >> 4;

  // positional encoding: lane fills row m = lane (sample base+lane), 64 cols
  {
    int m = lane;
    float t = near + ((float)(base + m) + 0.5f) * dt;
    float P[3] = {ox + t * dx, oy + t * dy, oz + t * dz};
    for (int c = 0; c < 64; c++) {
      float v;
      if (c < 3)        v = P[c];
      else if (c < 33) { int u = c - 3;  int i = u / 10, f = u - 10 * i; v = __sinf(ldexpf(P[i], f)); }
      else if (c < 63) { int u = c - 33; int i = u / 10, f = u - 10 * i; v = __cosf(ldexpf(P[i], f)); }
      else              v = 0.0f;
      bufA[m * ASTRIDE + c] = (_Float16)v;
    }
  }
  lds_fence();

  mlp_layer32(bufA, bufB, W1T, b1, 64, 256, lane, true);    // 63->256 relu
  lds_fence();
  mlp_layer32(bufB, bufA, W2T, b2, 256, 256, lane, true);   // 256->256 relu
  lds_fence();

  // build color input in bufB cols: [0..26]=dir_enc, [27..41]=fea, [42..63]=0
  // (in-wave LDS ops are in order, so overwriting bufB after layer2's reads is safe)
  {
    float D[3] = {dx, dy, dz};
    int m = lane;
    for (int c = 0; c < 27; c++) {
      float v;
      if (c < 3)        v = D[c];
      else if (c < 15) { int u = c - 3;  int i = u / 4, f = u - 4 * i; v = __sinf(ldexpf(D[i], f)); }
      else             { int u = c - 15; int i = u / 4, f = u - 4 * i; v = __cosf(ldexpf(D[i], f)); }
      bufB[m * ASTRIDE + c] = (_Float16)v;
    }
    for (int c = 42; c < 64; c++) bufB[m * ASTRIDE + c] = (_Float16)0.0f;
  }

  // geo layer: 256 -> 16 ; N=0 -> sigma (softplus), N=1..15 -> fea
  {
    v8f acc0 = {}, acc1 = {};
    for (int k0 = 0; k0 < 256; k0 += 32) {
      v16h B = load_b(W3T, 256, 0, k0, lane);
      acc0 = wmma_f16(load_a(bufA, 0,  k0, lane), B, acc0);
      acc1 = wmma_f16(load_a(bufA, 16, k0, lane), B, acc1);
    }
    int nn = lane & 15;
    float bv = b3[nn];
#pragma unroll
    for (int r = 0; r < 8; r++) {
      float x0 = acc0[r] + bv;
      float x1 = acc1[r] + bv;
      int row0 = r + hf * 8, row1 = 16 + r + hf * 8;
      if (nn == 0) {
        float s0 = fmaxf(x0, 0.0f) + log1pf(__expf(-fabsf(x0)));   // stable softplus
        float s1 = fmaxf(x1, 0.0f) + log1pf(__expf(-fabsf(x1)));
        rgbsig[(ray * N_SAMPLES + base + row0) * 4 + 3] = s0;
        rgbsig[(ray * N_SAMPLES + base + row1) * 4 + 3] = s1;
      } else {
        bufB[row0 * ASTRIDE + 26 + nn] = (_Float16)x0;
        bufB[row1 * ASTRIDE + 26 + nn] = (_Float16)x1;
      }
    }
  }
  lds_fence();

  mlp_layer32(bufB, bufA, Wc1T, bc1, 64, 64, lane, true);   // 42->64 relu
  lds_fence();

  // color out: 64 -> 3 (N padded to 16), sigmoid
  {
    v8f acc0 = {}, acc1 = {};
    for (int k0 = 0; k0 < 64; k0 += 32) {
      v16h B = load_b(Wc2T, 64, 0, k0, lane);
      acc0 = wmma_f16(load_a(bufA, 0,  k0, lane), B, acc0);
      acc1 = wmma_f16(load_a(bufA, 16, k0, lane), B, acc1);
    }
    int nn = lane & 15;
    if (nn < 3) {
      float bv = bc2[nn];
#pragma unroll
      for (int r = 0; r < 8; r++) {
        float x0 = acc0[r] + bv;
        float x1 = acc1[r] + bv;
        rgbsig[(ray * N_SAMPLES + base + r + hf * 8) * 4 + nn]      = 1.0f / (1.0f + __expf(-x0));
        rgbsig[(ray * N_SAMPLES + base + 16 + r + hf * 8) * 4 + nn] = 1.0f / (1.0f + __expf(-x1));
      }
    }
  }
}

// ---------------- per-ray compositing --------------------------------------
__global__ void composite_kernel(const float* __restrict__ rays_o,
                                 const float* __restrict__ rays_d,
                                 const float* __restrict__ rgbsig,
                                 float* __restrict__ out)
{
  int ray = blockIdx.x * blockDim.x + threadIdx.x;
  if (ray >= N_RAYS) return;
  float dx, dy, dz, ox, oy, oz, near, dt;
  ray_setup(rays_o, rays_d, ray, dx, dy, dz, ox, oy, oz, near, dt);
  float T = 1.0f, r = 0.0f, g = 0.0f, b = 0.0f, depth = 0.0f, wsum = 0.0f;
  const float4* rs = (const float4*)rgbsig;
  for (int i = 0; i < N_SAMPLES; i++) {
    float4 v = rs[ray * N_SAMPLES + i];
    float alpha = 1.0f - __expf(-v.w * dt);
    float w = (T > 1e-4f) ? alpha * T : 0.0f;      // T here == T_excl
    float t = near + ((float)i + 0.5f) * dt;
    r += w * v.x; g += w * v.y; b += w * v.z;
    depth += w * t; wsum += w;
    T *= (1.0f - alpha + 1e-10f);
  }
  float bg = 1.0f - wsum;
  out[ray * 3 + 0] = r + bg;
  out[ray * 3 + 1] = g + bg;
  out[ray * 3 + 2] = b + bg;
  out[N_RAYS * 3 + ray] = depth;
  out[N_RAYS * 4 + ray] = fminf(fmaxf(wsum, 1e-12f), 1000.0f);
}

// ---------------- launcher --------------------------------------------------
extern "C" void kernel_launch(void* const* d_in, const int* in_sizes, int n_in,
                              void* d_out, int out_size, void* d_ws, size_t ws_size,
                              hipStream_t stream)
{
  const float* rays_o = (const float*)d_in[0];
  const float* rays_d = (const float*)d_in[1];
  const float* W1  = (const float*)d_in[2];
  const float* b1  = (const float*)d_in[3];
  const float* W2  = (const float*)d_in[4];
  const float* b2  = (const float*)d_in[5];
  const float* W3  = (const float*)d_in[6];
  const float* b3  = (const float*)d_in[7];
  const float* Wc1 = (const float*)d_in[8];
  const float* bc1 = (const float*)d_in[9];
  const float* Wc2 = (const float*)d_in[10];
  const float* bc2 = (const float*)d_in[11];

  _Float16* wsh    = (_Float16*)d_ws;
  float*    rgbsig = (float*)((char*)d_ws + OFF_RGBSIG_BYTES);

  prepack_weights<<<(N_PACK + 255) / 256, 256, 0, stream>>>(W1, W2, W3, Wc1, Wc2, wsh);
  render_kernel<<<N_RAYS, 128, 4 * 2 * BUFHALFS * sizeof(_Float16), stream>>>(
      rays_o, rays_d, b1, b2, b3, bc1, bc2, wsh, rgbsig);
  composite_kernel<<<(N_RAYS + 255) / 256, 256, 0, stream>>>(rays_o, rays_d, rgbsig, (float*)d_out);
}
